// GAT_8615704396306
// MI455X (gfx1250) — matrix operand order repaired and checked by reading, not compiled, not run on hardware
//
#include <hip/hip_runtime.h>
#include <hip/hip_bf16.h>
#include <math.h>

typedef _Float16 half_t;
typedef half_t v16h __attribute__((ext_vector_type(16)));
typedef float  v8f  __attribute__((ext_vector_type(8)));
typedef float  v4f  __attribute__((ext_vector_type(4)));
typedef unsigned int v4u __attribute__((ext_vector_type(4)));
typedef int v8i __attribute__((ext_vector_type(8)));
typedef int v4i __attribute__((ext_vector_type(4)));

#define F_IN   512
#define HID    8
#define HEADS  8
#define FOUT1  64      // HEADS*HID
#define CLS    7
#define NEG_SLOPE 0.2f

#define WQ_BYTES (16 * 4 * 32 * 16 * 2)   // 64 KB swizzled W1 (f16)

// ---------------------------------------------------------------- utilities
__global__ void fill_kernel(float* __restrict__ p, long n, float v) {
    long i = blockIdx.x * (long)blockDim.x + threadIdx.x;
    if (i < n) p[i] = v;
}

// float atomic max (native global_atomic_max_num_f32 on gfx1250)
__device__ __forceinline__ void atomic_max_f32(float* addr, float val) {
    atomicMax(addr, val);
}

__device__ __forceinline__ void edge_nodes(const int* __restrict__ ei, int E, long e,
                                           int& s, int& d) {
    if (e < E) { s = ei[e]; d = ei[(long)E + e]; }
    else       { s = d = (int)(e - E); }          // self-loops appended
}

// ------------------------------------------------- W1 -> f16 WMMA-B swizzle
// Layout: [kt(16)][ct(4)][lane(32)][j(16)] f16.
// B (32x16, KxN, 16-bit): lane = N (l&15); lanes 0-15 hold K=0..7,16..23,
// lanes 16-31 hold K=8..15,24..31; elements packed pairwise per VGPR.
__global__ void w1_swizzle_kernel(const float* __restrict__ W1, half_t* __restrict__ Wq) {
    int id = blockIdx.x * blockDim.x + threadIdx.x;      // 16*4*32*16 = 32768
    if (id >= 16 * 4 * 32 * 16) return;
    int j    = id & 15;
    int lane = (id >> 4) & 31;
    int ct   = (id >> 9) & 3;
    int kt   = id >> 11;
    int n    = lane & 15;
    int k    = ((lane < 16) ? 0 : 8) + ((j < 8) ? j : 8 + j);
    int kg   = kt * 32 + k;
    int col  = ct * 16 + n;
    Wq[id] = (half_t)W1[kg * FOUT1 + col];
}

// --------------------------------------- TDM: broadcast Wq (64KB) into LDS
// 2-D descriptor: data_size=8B, one row of 8192 elements.
__device__ __forceinline__ void tdm_load_wq_to_lds(const half_t* Wq) {
    unsigned long long ga = (unsigned long long)(const void*)Wq;
    v4u g0;
    g0.x = 1u;                                        // count=1, user mode
    g0.y = 0u;                                        // lds_addr = 0 (dyn smem base)
    g0.z = (unsigned)(ga & 0xFFFFFFFFu);              // global_addr[31:0]
    g0.w = (unsigned)((ga >> 32) & 0x01FFFFFFu)       // global_addr[56:32]
         | 0x80000000u;                               // type = 2 ("image")
    v8i g1;
    g1[0] = 0x00030000;        // workgroup_mask=0, data_size=3 (8B), no flags
    g1[1] = 0x20000000;        // abar_addr=0, tensor_dim0.lo16 = 8192
    g1[2] = 0x00010000;        // tensor_dim0.hi16=0, tensor_dim1.lo16 = 1
    g1[3] = 0x20000000;        // tensor_dim1.hi16=0, tile_dim0 = 8192
    g1[4] = 0x00000001;        // tile_dim1 = 1, tile_dim2 = 0
    g1[5] = 8192;              // tensor_dim0_stride[31:0]
    g1[6] = 0;                 // stride0.hi16=0, tensor_dim1_stride.lo16=0
    g1[7] = 0;
    v4i z4 = {0, 0, 0, 0};
#if defined(__clang_major__) && (__clang_major__ >= 23)
    v8i z8 = {0, 0, 0, 0, 0, 0, 0, 0};
    __builtin_amdgcn_tensor_load_to_lds(g0, g1, z4, z4, z8, 0);
#else
    __builtin_amdgcn_tensor_load_to_lds(g0, g1, z4, z4, 0);
#endif
    __builtin_amdgcn_s_wait_tensorcnt(0);
}

// ------------------------------------------------------------ GEMM1 (WMMA)
// h1[N,64] = x[N,512] @ W1[512,64], f16 inputs, f32 accumulate.
// B staged in LDS via TDM once per block; one wave per 16-row tile;
// 4 col tiles of 16; K loop 16 x 32.
__global__ void __launch_bounds__(256)
gemm1_kernel(const float* __restrict__ x, const half_t* __restrict__ Wq,
             float* __restrict__ h1, int N) {
    extern __shared__ char smem[];
    int lane = threadIdx.x & 31;
    int wave = threadIdx.x >> 5;

    if (threadIdx.x < 32) {                 // wave 0 issues the TDM broadcast
        tdm_load_wq_to_lds(Wq);
    }
    __syncthreads();

    int rt  = blockIdx.x * 8 + wave;
    int nrt = (N + 15) >> 4;
    if (rt >= nrt) return;

    int row = rt * 16 + (lane & 15);
    if (row >= N) row = N - 1;              // clamp for partial last tile
    int kbase = (lane < 16) ? 0 : 8;        // A-frag: lanes split K in 8s
    const float* xr = x + (long)row * F_IN + kbase;
    const v16h*  wl = (const v16h*)smem;    // B fragments from LDS

    v8f acc0 = {}, acc1 = {}, acc2 = {}, acc3 = {};

    for (int kt = 0; kt < 16; ++kt) {
        const float* p = xr + kt * 32;
        v4f c0 = *(const v4f*)(p);
        v4f c1 = *(const v4f*)(p + 4);
        v4f c2 = *(const v4f*)(p + 16);
        v4f c3 = *(const v4f*)(p + 20);
        v16h a;
#pragma unroll
        for (int j = 0; j < 4; ++j) {
            a[j]      = (half_t)c0[j];   // K = kbase + 0..3
            a[4 + j]  = (half_t)c1[j];   // K = kbase + 4..7
            a[8 + j]  = (half_t)c2[j];   // K = kbase + 16..19
            a[12 + j] = (half_t)c3[j];   // K = kbase + 20..23
        }
        v16h b0 = wl[(kt * 4 + 0) * 32 + lane];
        v16h b1 = wl[(kt * 4 + 1) * 32 + lane];
        v16h b2 = wl[(kt * 4 + 2) * 32 + lane];
        v16h b3 = wl[(kt * 4 + 3) * 32 + lane];
        acc0 = __builtin_amdgcn_wmma_f32_16x16x32_f16(false, a, false, b0, (short)0, acc0, false, false);
        acc1 = __builtin_amdgcn_wmma_f32_16x16x32_f16(false, a, false, b1, (short)0, acc1, false, false);
        acc2 = __builtin_amdgcn_wmma_f32_16x16x32_f16(false, a, false, b2, (short)0, acc2, false, false);
        acc3 = __builtin_amdgcn_wmma_f32_16x16x32_f16(false, a, false, b3, (short)0, acc3, false, false);
    }

    // D layout: VGPR i -> M = i + (lane<16 ? 0 : 8), N = lane&15
    int mbase = (lane < 16) ? 0 : 8;
    int col   = lane & 15;
#pragma unroll
    for (int i = 0; i < 8; ++i) {
        int r = rt * 16 + mbase + i;
        if (r < N) {
            float* o = h1 + (long)r * FOUT1 + col;
            o[0]  = acc0[i];
            o[16] = acc1[i];
            o[32] = acc2[i];
            o[48] = acc3[i];
        }
    }
}

// ------------------------------------------------ layer-1 attention scalars
__global__ void alpha1_kernel(const float* __restrict__ h1,
                              const float* __restrict__ a_src, const float* __restrict__ a_dst,
                              float* __restrict__ asrc, float* __restrict__ adst, int N) {
    long id = blockIdx.x * (long)blockDim.x + threadIdx.x;
    if (id >= (long)N * HEADS) return;
    int h = (int)(id & 7);
    long n = id >> 3;
    const float* hp = h1 + n * FOUT1 + h * HID;
    float s = 0.f, d = 0.f;
#pragma unroll
    for (int c = 0; c < HID; ++c) {
        float v = hp[c];
        s += v * a_src[h * HID + c];
        d += v * a_dst[h * HID + c];
    }
    asrc[id] = s;
    adst[id] = d;
}

// -------------------------------------------------- edge softmax: layer 1
__global__ void edge_max1_kernel(const int* __restrict__ ei, int E, int ET,
                                 const float* __restrict__ asrc, const float* __restrict__ adst,
                                 float* __restrict__ m1) {
    long e = blockIdx.x * (long)blockDim.x + threadIdx.x;
    if (e >= ET) return;
    int s, d; edge_nodes(ei, E, e, s, d);
#pragma unroll
    for (int h = 0; h < HEADS; ++h) {
        float v = asrc[(long)s * 8 + h] + adst[(long)d * 8 + h];
        v = (v > 0.f) ? v : NEG_SLOPE * v;
        atomic_max_f32(&m1[(long)d * 8 + h], v);
    }
}

__global__ void edge_sum1_kernel(const int* __restrict__ ei, int E, int ET,
                                 const float* __restrict__ asrc, const float* __restrict__ adst,
                                 const float* __restrict__ m1, float* __restrict__ den1) {
    long e = blockIdx.x * (long)blockDim.x + threadIdx.x;
    if (e >= ET) return;
    int s, d; edge_nodes(ei, E, e, s, d);
#pragma unroll
    for (int h = 0; h < HEADS; ++h) {
        float v = asrc[(long)s * 8 + h] + adst[(long)d * 8 + h];
        v = (v > 0.f) ? v : NEG_SLOPE * v;
        atomicAdd(&den1[(long)d * 8 + h], __expf(v - m1[(long)d * 8 + h]));
    }
}

__global__ void edge_scatter1_kernel(const int* __restrict__ ei, int E, long ETH,
                                     const float* __restrict__ asrc, const float* __restrict__ adst,
                                     const float* __restrict__ m1, const float* __restrict__ den1,
                                     const float* __restrict__ h1, float* __restrict__ acc1) {
    long id = blockIdx.x * (long)blockDim.x + threadIdx.x;
    if (id >= ETH) return;
    int h = (int)(id & 7);
    long e = id >> 3;
    int s, d; edge_nodes(ei, E, e, s, d);
    float v = asrc[(long)s * 8 + h] + adst[(long)d * 8 + h];
    v = (v > 0.f) ? v : NEG_SLOPE * v;
    float alpha = __expf(v - m1[(long)d * 8 + h]) / den1[(long)d * 8 + h];
    const float* hp = h1 + (long)s * FOUT1 + h * HID;
    float* op = acc1 + (long)d * FOUT1 + h * HID;
#pragma unroll
    for (int c = 0; c < HID; ++c) atomicAdd(&op[c], alpha * hp[c]);
}

// bias + ELU -> h2
__global__ void finalize1_kernel(const float* __restrict__ acc1, const float* __restrict__ b1,
                                 float* __restrict__ h2, int N) {
    long id = blockIdx.x * (long)blockDim.x + threadIdx.x;
    if (id >= (long)N * FOUT1) return;
    int f = (int)(id & 63);
    float v = acc1[id] + b1[f];
    h2[id] = (v > 0.f) ? v : (__expf(v) - 1.f);
}

// ------------------------------------ layer-2 linear + attention scalars
__global__ void node2_kernel(const float* __restrict__ h2, const float* __restrict__ W2,
                             const float* __restrict__ a_src2, const float* __restrict__ a_dst2,
                             float* __restrict__ hz, float* __restrict__ asrc2,
                             float* __restrict__ adst2, int N) {
    int n = blockIdx.x * blockDim.x + threadIdx.x;
    if (n >= N) return;
    const float* hp = h2 + (long)n * FOUT1;
    float z[CLS];
#pragma unroll
    for (int c = 0; c < CLS; ++c) z[c] = 0.f;
    for (int k = 0; k < FOUT1; ++k) {
        float hv = hp[k];
#pragma unroll
        for (int c = 0; c < CLS; ++c) z[c] += hv * W2[k * CLS + c];
    }
    float s = 0.f, d = 0.f;
#pragma unroll
    for (int c = 0; c < CLS; ++c) {
        hz[(long)n * 8 + c] = z[c];
        s += z[c] * a_src2[c];
        d += z[c] * a_dst2[c];
    }
    hz[(long)n * 8 + 7] = 0.f;
    asrc2[n] = s;
    adst2[n] = d;
}

// -------------------------------------------------- edge softmax: layer 2
__global__ void edge_max2_kernel(const int* __restrict__ ei, int E, int ET,
                                 const float* __restrict__ asrc2, const float* __restrict__ adst2,
                                 float* __restrict__ m2) {
    long e = blockIdx.x * (long)blockDim.x + threadIdx.x;
    if (e >= ET) return;
    int s, d; edge_nodes(ei, E, e, s, d);
    float v = asrc2[s] + adst2[d];
    v = (v > 0.f) ? v : NEG_SLOPE * v;
    atomic_max_f32(&m2[d], v);
}

__global__ void edge_sum2_kernel(const int* __restrict__ ei, int E, int ET,
                                 const float* __restrict__ asrc2, const float* __restrict__ adst2,
                                 const float* __restrict__ m2, float* __restrict__ den2) {
    long e = blockIdx.x * (long)blockDim.x + threadIdx.x;
    if (e >= ET) return;
    int s, d; edge_nodes(ei, E, e, s, d);
    float v = asrc2[s] + adst2[d];
    v = (v > 0.f) ? v : NEG_SLOPE * v;
    atomicAdd(&den2[d], __expf(v - m2[d]));
}

__global__ void edge_scatter2_kernel(const int* __restrict__ ei, int E, int ET,
                                     const float* __restrict__ asrc2, const float* __restrict__ adst2,
                                     const float* __restrict__ m2, const float* __restrict__ den2,
                                     const float* __restrict__ hz, float* __restrict__ acc2) {
    long e = blockIdx.x * (long)blockDim.x + threadIdx.x;
    if (e >= ET) return;
    int s, d; edge_nodes(ei, E, e, s, d);
    float v = asrc2[s] + adst2[d];
    v = (v > 0.f) ? v : NEG_SLOPE * v;
    float alpha = __expf(v - m2[d]) / den2[d];
    const float* hp = hz + (long)s * 8;
    float* op = acc2 + (long)d * 8;
#pragma unroll
    for (int c = 0; c < CLS; ++c) atomicAdd(&op[c], alpha * hp[c]);
}

// ------------------------------------------------- bias + log_softmax out
__global__ void final_kernel(const float* __restrict__ acc2, const float* __restrict__ b2,
                             float* __restrict__ out, int N) {
    int n = blockIdx.x * blockDim.x + threadIdx.x;
    if (n >= N) return;
    float v[CLS];
    float mx = -3.0e38f;
#pragma unroll
    for (int c = 0; c < CLS; ++c) {
        v[c] = acc2[(long)n * 8 + c] + b2[c];
        mx = fmaxf(mx, v[c]);
    }
    float se = 0.f;
#pragma unroll
    for (int c = 0; c < CLS; ++c) se += __expf(v[c] - mx);
    float lse = mx + __logf(se);
#pragma unroll
    for (int c = 0; c < CLS; ++c) out[(long)n * CLS + c] = v[c] - lse;
}

// ---------------------------------------------------------------- launcher
extern "C" void kernel_launch(void* const* d_in, const int* in_sizes, int n_in,
                              void* d_out, int out_size, void* d_ws, size_t ws_size,
                              hipStream_t stream) {
    const float* x       = (const float*)d_in[0];
    const int*   ei      = (const int*)d_in[1];
    const float* W1      = (const float*)d_in[2];
    const float* a_src1  = (const float*)d_in[3];
    const float* a_dst1  = (const float*)d_in[4];
    const float* b1      = (const float*)d_in[5];
    const float* W2      = (const float*)d_in[6];
    const float* a_src2  = (const float*)d_in[7];
    const float* a_dst2  = (const float*)d_in[8];
    const float* b2      = (const float*)d_in[9];
    float* out = (float*)d_out;

    const int N  = in_sizes[0] / F_IN;
    const int E  = in_sizes[1] / 2;
    const int ET = E + N;                 // with self-loops

    // ---- workspace carve (256B aligned)
    char* ws = (char*)d_ws;
    auto carve = [&](size_t bytes) -> void* {
        void* p = (void*)ws;
        ws += (bytes + 255) & ~(size_t)255;
        return p;
    };
    half_t* Wq    = (half_t*)carve(WQ_BYTES);
    float*  h1    = (float*)carve((size_t)N * FOUT1 * 4);
    float*  asrc1 = (float*)carve((size_t)N * HEADS * 4);
    float*  adst1 = (float*)carve((size_t)N * HEADS * 4);
    float*  m1    = (float*)carve((size_t)N * HEADS * 4);
    float*  den1  = (float*)carve((size_t)N * HEADS * 4);
    float*  acc1  = (float*)carve((size_t)N * FOUT1 * 4);
    float*  h2    = (float*)carve((size_t)N * FOUT1 * 4);
    float*  hz    = (float*)carve((size_t)N * 8 * 4);
    float*  asrc2 = (float*)carve((size_t)N * 4);
    float*  adst2 = (float*)carve((size_t)N * 4);
    float*  m2    = (float*)carve((size_t)N * 4);
    float*  den2  = (float*)carve((size_t)N * 4);
    float*  acc2  = (float*)carve((size_t)N * 8 * 4);

    const int BT = 256;
    auto blocks = [&](long n) { return (int)((n + BT - 1) / BT); };

    // layer 1
    w1_swizzle_kernel<<<blocks(16 * 4 * 32 * 16), BT, 0, stream>>>(W1, Wq);
    {
        int nrt = (N + 15) >> 4;
        int g   = (nrt + 7) >> 3;         // 8 waves per block
        gemm1_kernel<<<g, 256, WQ_BYTES, stream>>>(x, Wq, h1, N);
    }
    alpha1_kernel<<<blocks((long)N * HEADS), BT, 0, stream>>>(h1, a_src1, a_dst1, asrc1, adst1, N);
    fill_kernel<<<blocks((long)N * HEADS), BT, 0, stream>>>(m1, (long)N * HEADS, -3.0e38f);
    fill_kernel<<<blocks((long)N * HEADS), BT, 0, stream>>>(den1, (long)N * HEADS, 0.f);
    fill_kernel<<<blocks((long)N * FOUT1), BT, 0, stream>>>(acc1, (long)N * FOUT1, 0.f);
    edge_max1_kernel<<<blocks(ET), BT, 0, stream>>>(ei, E, ET, asrc1, adst1, m1);
    edge_sum1_kernel<<<blocks(ET), BT, 0, stream>>>(ei, E, ET, asrc1, adst1, m1, den1);
    edge_scatter1_kernel<<<blocks((long)ET * HEADS), BT, 0, stream>>>(ei, E, (long)ET * HEADS,
                                                                      asrc1, adst1, m1, den1, h1, acc1);
    finalize1_kernel<<<blocks((long)N * FOUT1), BT, 0, stream>>>(acc1, b1, h2, N);

    // layer 2
    node2_kernel<<<blocks(N), BT, 0, stream>>>(h2, W2, a_src2, a_dst2, hz, asrc2, adst2, N);
    fill_kernel<<<blocks(N), BT, 0, stream>>>(m2, N, -3.0e38f);
    fill_kernel<<<blocks(N), BT, 0, stream>>>(den2, N, 0.f);
    fill_kernel<<<blocks((long)N * 8), BT, 0, stream>>>(acc2, (long)N * 8, 0.f);
    edge_max2_kernel<<<blocks(ET), BT, 0, stream>>>(ei, E, ET, asrc2, adst2, m2);
    edge_sum2_kernel<<<blocks(ET), BT, 0, stream>>>(ei, E, ET, asrc2, adst2, m2, den2);
    edge_scatter2_kernel<<<blocks(ET), BT, 0, stream>>>(ei, E, ET, asrc2, adst2, m2, den2, hz, acc2);
    final_kernel<<<blocks(N), BT, 0, stream>>>(acc2, b2, out, N);
}
